// SmyrfAttention_20942260535709
// MI455X (gfx1250) — matrix operand order repaired and checked by reading, not compiled, not run on hardware
//
#include <hip/hip_runtime.h>
#include <hip/hip_bf16.h>
#include <math.h>

// ---------------------------------------------------------------------------
// SMYRF attention for MI455X (gfx1250, wave32, WMMA).
//
// B=2, T=8192, H=8, E=64, bs=B*H=16, N_HASHES=4, cluster=128, nc=64.
// ~17 GFLOP vs hundreds of MB of HBM traffic @ 23.3 TB/s => memory bound.
// Cluster attention runs in LDS with v_wmma_f32_16x16x32_f16 (f32 accum);
// argsort = in-LDS bitonic sort of (orderable-float<<32 | index) u64 keys
// (stable ascending, matching jnp.argsort).  NOTE: alpha is generated with a
// deterministic on-device splitmix32 + Box-Muller generator; JAX
// threefry("key 42") is not bit-reproducible here.  beta is omitted: a
// per-hash constant shift cannot change argsort order.
// ---------------------------------------------------------------------------

typedef __attribute__((ext_vector_type(16))) _Float16 v16h;
typedef __attribute__((ext_vector_type(8)))  float    v8f;
typedef __attribute__((ext_vector_type(4)))  _Float16 h4;

constexpr int Bb   = 2;
constexpr int Hh   = 8;
constexpr int BSZ  = Bb * Hh;   // 16
constexpr int T    = 8192;
constexpr int E    = 64;
constexpr int NH   = 4;
constexpr int CL   = 128;       // cluster (Q_ATTN == K_ATTN)
constexpr int NC   = T / CL;    // 64

constexpr int QSTR = 72;        // f16 row stride for Q/K/V tiles (bank pad)
constexpr int SSTR = 132;       // f32 row stride for S
constexpr int PSTR = 2 * SSTR;  // f16 row stride for P (aliases S bytes)
constexpr int OSTR = 68;        // f32 row stride for O (aliases Qh/Kh bytes)

constexpr size_t ATTN_SMEM = 2 * CL * 4             // iq, ik
                           + (size_t)CL * SSTR * 4  // S (reused as P)
                           + 3 * (size_t)CL * QSTR * 2; // Qh, Kh, Vh (O overlays Qh+Kh)
constexpr size_t SORT_SMEM = (size_t)T * 8;

static_assert((size_t)CL * OSTR * 4 <= 2 * (size_t)CL * QSTR * 2,
              "O staging must fit in Qh+Kh region");

// q/k/v global layout is (B, T, H, E); bs = b*H + h.
__device__ __forceinline__ size_t gidx(int bs, int t, int e) {
  int b = bs >> 3, hh = bs & 7;
  return (((size_t)b * T + t) * Hh + hh) * E + e;
}

__device__ __forceinline__ unsigned splitmix32(unsigned x) {
  x += 0x9e3779b9u; x ^= x >> 16; x *= 0x85ebca6bu;
  x ^= x >> 13;     x *= 0xc2b2ae35u; x ^= x >> 16;
  return x;
}
__device__ __forceinline__ float u01(unsigned x) {
  return (float)(x >> 8) * (1.0f / 16777216.0f) + (0.5f / 16777216.0f);
}
__device__ __forceinline__ float gauss_hash(unsigned idx) {
  float u1 = u01(splitmix32(2u * idx + 1u));
  float u2 = u01(splitmix32(2u * idx + 2u));
  return sqrtf(-2.0f * __logf(u1)) * __cosf(6.28318530718f * u2);
}
__device__ __forceinline__ unsigned long long packkey(float f, unsigned idx) {
  unsigned u = __float_as_uint(f);
  u = (u & 0x80000000u) ? ~u : (u | 0x80000000u);   // total order ascending
  return ((unsigned long long)u << 32) | idx;        // index ties -> stable
}

// --- WMMA fragment loaders (f16, 16x16x32) --------------------------------
// A (16x32): lane = row (lane&15); halves j=0..15 -> K = 16*(lane>>4)+j.
__device__ __forceinline__ v16h load_A(const _Float16* M, int stride,
                                       int row0, int k0, int lane) {
  int r  = row0 + (lane & 15);
  int kb = (lane >> 4) << 4;
  const _Float16* p0 = M + (size_t)r * stride + k0 + kb;
  const _Float16* p1 = p0 + 8;
  v16h a;
#pragma unroll
  for (int j = 0; j < 8; ++j) { a[j] = p0[j]; a[8 + j] = p1[j]; }
  return a;
}
// B (32x16): lane = column (lane&15); halves j=0..15 -> K = 16*(lane>>4)+j.
// Variant 1: B = Mᵀ, M row-major [N x K] (Q·Kᵀ: M = K-matrix).
__device__ __forceinline__ v16h load_B_T(const _Float16* M, int stride,
                                         int col0, int k0, int lane) {
  int n  = col0 + (lane & 15);
  int kb = k0 + ((lane >> 4) << 4);
  const _Float16* p = M + (size_t)n * stride + kb;   // contiguous 16 halves
  v16h b;
#pragma unroll
  for (int j = 0; j < 16; ++j) b[j] = p[j];
  return b;
}
// Variant 2: B = M, M row-major [K x N] (P·V: M = V).
__device__ __forceinline__ v16h load_B(const _Float16* M, int stride,
                                       int k0, int n0, int lane) {
  int n  = n0 + (lane & 15);
  int kb = k0 + ((lane >> 4) << 4);
  v16h b;
#pragma unroll
  for (int j = 0; j < 16; ++j) b[j] = M[(size_t)(kb + j) * stride + n];
  return b;
}

// ---------------------------------------------------------------------------
__global__ void smyrf_init_kernel(unsigned* MQ2, unsigned* MK2) {
  int t = threadIdx.x;
  if (t < BSZ) { MQ2[t] = 0u; MK2[t] = 0u; }
}

// Fused single pass over q,k: row norm^2 (+ per-bs max via atomicMax on
// float bits, valid for non-negative values) and the four alpha-dots.
__global__ void smyrf_stats_kernel(const float* __restrict__ q,
                                   const float* __restrict__ k,
                                   float* __restrict__ qn2,
                                   float* __restrict__ kn2,
                                   float* __restrict__ qdot,   // [row*4+h]
                                   float* __restrict__ kdot,
                                   unsigned* MQ2, unsigned* MK2) {
  __shared__ float alphaS[E * NH];            // alpha[e][h] at e*NH+h
  int tid = threadIdx.x;
  if (tid < E * NH) alphaS[tid] = gauss_hash((unsigned)tid);
  __syncthreads();

  int r = blockIdx.x * blockDim.x + tid;
  if (r >= BSZ * T) return;
  int bs = r / T, t = r % T;

  float sq = 0.f, sk = 0.f;
  float aq[NH] = {0.f, 0.f, 0.f, 0.f};
  float ak[NH] = {0.f, 0.f, 0.f, 0.f};
  for (int e4 = 0; e4 < E; e4 += 4) {
    float4 vq = *(const float4*)(q + gidx(bs, t, e4));
    float4 vk = *(const float4*)(k + gidx(bs, t, e4));
    float xq[4] = {vq.x, vq.y, vq.z, vq.w};
    float xk[4] = {vk.x, vk.y, vk.z, vk.w};
#pragma unroll
    for (int u = 0; u < 4; ++u) {
      sq = fmaf(xq[u], xq[u], sq);
      sk = fmaf(xk[u], xk[u], sk);
#pragma unroll
      for (int h = 0; h < NH; ++h) {
        aq[h] = fmaf(xq[u], alphaS[(e4 + u) * NH + h], aq[h]);
        ak[h] = fmaf(xk[u], alphaS[(e4 + u) * NH + h], ak[h]);
      }
    }
  }
  qn2[r] = sq; kn2[r] = sk;
#pragma unroll
  for (int h = 0; h < NH; ++h) {
    qdot[(size_t)r * NH + h] = aq[h];
    kdot[(size_t)r * NH + h] = ak[h];
  }
  atomicMax(MQ2 + bs, __float_as_uint(sq));
  atomicMax(MK2 + bs, __float_as_uint(sk));
}

// XBOX+ extension term + sortable-key pack (tiny pass, no q/k reads).
__global__ void smyrf_pack_kernel(const float* __restrict__ qn2,
                                  const float* __restrict__ kn2,
                                  const float* __restrict__ qdot,
                                  const float* __restrict__ kdot,
                                  const unsigned* __restrict__ MQ2,
                                  const unsigned* __restrict__ MK2,
                                  unsigned long long* __restrict__ qkeys,
                                  unsigned long long* __restrict__ kkeys) {
  int r = blockIdx.x * blockDim.x + threadIdx.x;
  if (r >= BSZ * T) return;
  int bs = r / T, t = r % T;
  float qext = sqrtf(fmaxf(__uint_as_float(MQ2[bs]) - qn2[r], 0.f));
  float kext = sqrtf(fmaxf(__uint_as_float(MK2[bs]) - kn2[r], 0.f));
#pragma unroll
  for (int h = 0; h < NH; ++h) {
    float qh = qdot[(size_t)r * NH + h] + qext * gauss_hash(E * NH + h);
    float kh = kdot[(size_t)r * NH + h] + kext * gauss_hash((E + 1) * NH + h);
    qkeys[((size_t)h * BSZ + bs) * T + t] = packkey(qh, (unsigned)t);
    kkeys[((size_t)h * BSZ + bs) * T + t] = packkey(kh, (unsigned)t);
  }
}

// One workgroup per (hash, bs) segment; full 8192-element bitonic sort in LDS.
__global__ void smyrf_sort_kernel(const unsigned long long* __restrict__ qkeys,
                                  const unsigned long long* __restrict__ kkeys,
                                  unsigned* __restrict__ qpos,
                                  unsigned* __restrict__ kpos) {
  extern __shared__ unsigned long long sk[];
  int seg = blockIdx.x;
  const unsigned long long* src;
  unsigned* dst;
  if (seg < NH * BSZ) { src = qkeys + (size_t)seg * T; dst = qpos + (size_t)seg * T; }
  else { int s2 = seg - NH * BSZ; src = kkeys + (size_t)s2 * T; dst = kpos + (size_t)s2 * T; }

  int tid = threadIdx.x;
  for (int i = tid; i < T; i += 1024) sk[i] = src[i];
  __syncthreads();

  for (unsigned ksz = 2; ksz <= (unsigned)T; ksz <<= 1) {
    for (unsigned j = ksz >> 1; j > 0; j >>= 1) {
      for (unsigned i = tid; i < (unsigned)T; i += 1024) {
        unsigned ixj = i ^ j;
        if (ixj > i) {
          unsigned long long a = sk[i], b = sk[ixj];
          bool up = ((i & ksz) == 0);
          if ((a > b) == up) { sk[i] = b; sk[ixj] = a; }
        }
      }
      __syncthreads();
    }
  }
  for (int i = tid; i < T; i += 1024) dst[i] = (unsigned)(sk[i] & 0xFFFFFFFFu);
}

// One workgroup (256 thr = 8 waves) per (hash, bs, cluster).
__global__ void smyrf_attn_kernel(const float* __restrict__ q,
                                  const float* __restrict__ k,
                                  const float* __restrict__ v,
                                  const unsigned* __restrict__ qpos,
                                  const unsigned* __restrict__ kpos,
                                  float* __restrict__ bo,
                                  float* __restrict__ lse) {
  extern __shared__ unsigned char smem[];
  unsigned* iq = (unsigned*)smem;
  unsigned* ik = iq + CL;
  float*    S  = (float*)(ik + CL);
  _Float16* Qh = (_Float16*)(S + (size_t)CL * SSTR);
  _Float16* Kh = Qh + (size_t)CL * QSTR;
  _Float16* Vh = Kh + (size_t)CL * QSTR;
  _Float16* P  = (_Float16*)S;     // aliases S bytes (safe in-place rewrite)
  float*    O  = (float*)Qh;       // aliases Qh+Kh (dead after QK phase)

  int blk = blockIdx.x;
  int c   = blk % NC;
  int hb  = blk / NC;              // hash*BSZ + bs
  int bs  = hb % BSZ;
  int tid  = threadIdx.x;
  int lane = tid & 31;
  int wave = tid >> 5;

  if (tid < CL) {
    iq[tid] = qpos[(size_t)hb * T + c * CL + tid];
    ik[tid] = kpos[(size_t)hb * T + c * CL + tid];
  }
  __syncthreads();

  // Gather + f32->f16 convert: 3 x 128x64 tiles, float4 loads (b128).
  for (int i = tid; i < CL * (E / 4); i += 256) {
    int r  = i >> 4;
    int e4 = (i & 15) << 2;
    float4 vq = *(const float4*)(q + gidx(bs, (int)iq[r], e4));
    float4 vk = *(const float4*)(k + gidx(bs, (int)ik[r], e4));
    float4 vv = *(const float4*)(v + gidx(bs, (int)ik[r], e4));
    h4 hq = { (_Float16)vq.x, (_Float16)vq.y, (_Float16)vq.z, (_Float16)vq.w };
    h4 hk = { (_Float16)vk.x, (_Float16)vk.y, (_Float16)vk.z, (_Float16)vk.w };
    h4 hv = { (_Float16)vv.x, (_Float16)vv.y, (_Float16)vv.z, (_Float16)vv.w };
    *(h4*)(Qh + (size_t)r * QSTR + e4) = hq;
    *(h4*)(Kh + (size_t)r * QSTR + e4) = hk;
    *(h4*)(Vh + (size_t)r * QSTR + e4) = hv;
  }
  __syncthreads();

  // S = Q Kᵀ : wave w owns M-tile w; 8 N-tiles, K=64 in 2 steps of 32.
  int mt = wave;
#pragma unroll
  for (int nt = 0; nt < 8; ++nt) {
    v8f acc = {0.f, 0.f, 0.f, 0.f, 0.f, 0.f, 0.f, 0.f};
#pragma unroll
    for (int kt = 0; kt < 2; ++kt) {
      v16h a = load_A(Qh, QSTR, mt * 16, kt * 32, lane);
      v16h b = load_B_T(Kh, QSTR, nt * 16, kt * 32, lane);
      acc = __builtin_amdgcn_wmma_f32_16x16x32_f16(false, a, false, b,
                                                   (short)0, acc, false, false);
    }
    int col = nt * 16 + (lane & 15);
    int r0  = mt * 16 + ((lane >> 4) << 3);
#pragma unroll
    for (int r = 0; r < 8; ++r) S[(size_t)(r0 + r) * SSTR + col] = acc[r];
  }
  __syncthreads();

  // Row softmax; P (f16) written in place over S (ascending col => the 2-byte
  // write at 2c only lands on already-consumed 4-byte element c/2).
  if (tid < CL) {
    float* srow = S + (size_t)tid * SSTR;
    _Float16* prow = P + (size_t)tid * PSTR;
    float m = -3.4e38f;
    for (int cc = 0; cc < CL; ++cc) m = fmaxf(m, srow[cc]);
    float sum = 0.f;
    for (int cc = 0; cc < CL; ++cc) {
      float ev = __expf(srow[cc] - m);
      sum += ev;
      prow[cc] = (_Float16)ev;
    }
    float inv = 1.0f / sum;
    for (int cc = 0; cc < CL; ++cc) prow[cc] = (_Float16)((float)prow[cc] * inv);
    lse[(size_t)hb * T + iq[tid]] = m + __logf(sum);   // scatter to orig pos
  }
  __syncthreads();

  // O = P V : N=64 in 4 tiles; K=128 in 4 steps of 32.  Stage O in LDS
  // (overlays dead Qh/Kh), then store coalesced float4 rows scattered by q_pos.
#pragma unroll
  for (int nt = 0; nt < 4; ++nt) {
    v8f acc = {0.f, 0.f, 0.f, 0.f, 0.f, 0.f, 0.f, 0.f};
#pragma unroll
    for (int kt = 0; kt < 4; ++kt) {
      v16h a = load_A(P, PSTR, mt * 16, kt * 32, lane);
      v16h b = load_B(Vh, QSTR, kt * 32, nt * 16, lane);
      acc = __builtin_amdgcn_wmma_f32_16x16x32_f16(false, a, false, b,
                                                   (short)0, acc, false, false);
    }
    int col = nt * 16 + (lane & 15);
    int r0  = mt * 16 + ((lane >> 4) << 3);
#pragma unroll
    for (int r = 0; r < 8; ++r) O[(size_t)(r0 + r) * OSTR + col] = acc[r];
  }
  __syncthreads();

  for (int i = tid; i < CL * (E / 4); i += 256) {
    int r  = i >> 4;
    int e4 = (i & 15) << 2;
    float4 o4 = *(const float4*)(O + (size_t)r * OSTR + e4);
    *(float4*)(bo + ((size_t)hb * T + iq[r]) * E + e4) = o4;
  }
}

// Cross-hash logsumexp merge -> out (B, T, H, E); float4 per thread.
__global__ void smyrf_merge_kernel(const float* __restrict__ bo,
                                   const float* __restrict__ lse,
                                   float* __restrict__ out) {
  size_t gid = (size_t)blockIdx.x * blockDim.x + threadIdx.x;
  if (gid >= (size_t)BSZ * T * (E / 4)) return;
  int e  = (int)(gid & 15) << 2;
  int t  = (int)((gid >> 4) & (T - 1));
  int bs = (int)(gid >> 17);

  float l[NH];
  float m = -3.4e38f;
#pragma unroll
  for (int h = 0; h < NH; ++h) {
    l[h] = lse[((size_t)h * BSZ + bs) * T + t];
    m = fmaxf(m, l[h]);
  }
  float s = 0.f, w[NH];
#pragma unroll
  for (int h = 0; h < NH; ++h) { w[h] = __expf(l[h] - m); s += w[h]; }
  float inv = 1.0f / s;

  float4 o = {0.f, 0.f, 0.f, 0.f};
#pragma unroll
  for (int h = 0; h < NH; ++h) {
    float wh = w[h] * inv;
    float4 x = *(const float4*)(bo + (((size_t)h * BSZ + bs) * T + t) * E + e);
    o.x = fmaf(wh, x.x, o.x); o.y = fmaf(wh, x.y, o.y);
    o.z = fmaf(wh, x.z, o.z); o.w = fmaf(wh, x.w, o.w);
  }
  int b = bs >> 3, hh = bs & 7;
  *(float4*)(out + (((size_t)b * T + t) * Hh + hh) * E + e) = o;
}

// ---------------------------------------------------------------------------
extern "C" void kernel_launch(void* const* d_in, const int* in_sizes, int n_in,
                              void* d_out, int out_size, void* d_ws, size_t ws_size,
                              hipStream_t stream) {
  (void)in_sizes; (void)n_in; (void)out_size; (void)ws_size;
  const float* q = (const float*)d_in[0];
  const float* k = (const float*)d_in[1];
  const float* v = (const float*)d_in[2];
  float* out = (float*)d_out;

  char* ws = (char*)d_ws;
  size_t off = 0;
  auto carve = [&](size_t bytes) -> void* {
    void* p = ws + off;
    off += (bytes + 255) & ~(size_t)255;
    return p;
  };
  float*    qn2   = (float*)carve((size_t)BSZ * T * 4);
  float*    kn2   = (float*)carve((size_t)BSZ * T * 4);
  float*    qdot  = (float*)carve((size_t)BSZ * T * NH * 4);
  float*    kdot  = (float*)carve((size_t)BSZ * T * NH * 4);
  unsigned* MQ2   = (unsigned*)carve(256);
  unsigned* MK2   = (unsigned*)carve(256);
  unsigned long long* qkeys = (unsigned long long*)carve((size_t)NH * BSZ * T * 8);
  unsigned long long* kkeys = (unsigned long long*)carve((size_t)NH * BSZ * T * 8);
  unsigned* qpos  = (unsigned*)carve((size_t)NH * BSZ * T * 4);
  unsigned* kpos  = (unsigned*)carve((size_t)NH * BSZ * T * 4);
  float*    lse   = (float*)carve((size_t)NH * BSZ * T * 4);
  float*    bo    = (float*)carve((size_t)NH * BSZ * T * E * 4);

  hipFuncSetAttribute((const void*)smyrf_sort_kernel,
                      hipFuncAttributeMaxDynamicSharedMemorySize, (int)SORT_SMEM);
  hipFuncSetAttribute((const void*)smyrf_attn_kernel,
                      hipFuncAttributeMaxDynamicSharedMemorySize, (int)ATTN_SMEM);

  smyrf_init_kernel<<<1, 64, 0, stream>>>(MQ2, MK2);
  smyrf_stats_kernel<<<(BSZ * T) / 256, 256, 0, stream>>>(q, k, qn2, kn2,
                                                          qdot, kdot, MQ2, MK2);
  smyrf_pack_kernel<<<(BSZ * T) / 256, 256, 0, stream>>>(qn2, kn2, qdot, kdot,
                                                         MQ2, MK2, qkeys, kkeys);
  smyrf_sort_kernel<<<2 * NH * BSZ, 1024, SORT_SMEM, stream>>>(qkeys, kkeys, qpos, kpos);
  smyrf_attn_kernel<<<NH * BSZ * NC, 256, ATTN_SMEM, stream>>>(q, k, v, qpos, kpos, bo, lse);
  smyrf_merge_kernel<<<(BSZ * T * (E / 4)) / 256, 256, 0, stream>>>(bo, lse, out);
}
